// AttnDecoder_28286654612081
// MI455X (gfx1250) — compile-verified, matrix operand order checked by reading, not run on hardware
//
#include <hip/hip_runtime.h>
#include <hip/hip_bf16.h>
#include <math.h>

// Problem dims (fixed by the reference)
static constexpr int Bq = 32;     // batch
static constexpr int Sq = 128;    // source length
static constexpr int Hq = 1024;   // hidden
static constexpr int Vq = 32000;  // vocab
static constexpr int Tq = 32;     // decode steps

typedef __attribute__((ext_vector_type(16))) __bf16 v16bf;
typedef __attribute__((ext_vector_type(8)))  float  v8f;

__device__ __forceinline__ unsigned short f2bf(float f) {
  unsigned int u = __float_as_uint(f);
  u += 0x7FFFu + ((u >> 16) & 1u);   // round-to-nearest-even
  return (unsigned short)(u >> 16);
}

// ---------------------------------------------------------------- cast f32->bf16
__global__ void cast_f32_bf16_k(const float* __restrict__ src,
                                unsigned short* __restrict__ dst, int n) {
  int i = blockIdx.x * blockDim.x + threadIdx.x;
  int stride = gridDim.x * blockDim.x;
  for (; i < n; i += stride) dst[i] = f2bf(src[i]);
}

// ---------------------------------------------------------------- embedding gather (B,T,H) bf16
__global__ void gather_emb_k(const float* __restrict__ emb, const int* __restrict__ tgt,
                             unsigned short* __restrict__ out) {
  int i = blockIdx.x * blockDim.x + threadIdx.x;   // over B*T*H
  if (i >= Bq * Tq * Hq) return;
  int h  = i % Hq;
  int bt = i / Hq;
  int t  = bt % Tq;
  int b  = bt / Tq;
  int tok = (t == 0) ? 0 : tgt[b * Tq + t - 1];    // SOS = 0, then teacher forcing
  out[i] = f2bf(emb[(size_t)tok * Hq + h]);
}

// ---------------------------------------------------------------- WMMA GEMM:
// C(MxN,f32) = A(MxK,bf16,row-major,lda) * B(NxK,bf16,row-major,ldb)^T + bias
// Each wave computes TWO adjacent 16x16 output tiles (16x32 of C), reusing the
// A fragment across both WMMAs (3 b128 loads per v_wmma instead of 4).
// 8 waves per block -> 256 output columns per block. Requires N % 256 == 0,
// M % 16 == 0 (true for all GEMMs here: N in {1024, 3072, 32000}).
__global__ void wmma_gemm_bf16_k(const unsigned short* __restrict__ A, int lda,
                                 const unsigned short* __restrict__ Bm, int ldb,
                                 const float* __restrict__ bias,
                                 float* __restrict__ C, unsigned long long ldc,
                                 unsigned short* __restrict__ Cbf, int ldcbf,
                                 int M, int N, int K) {
  const int wave  = threadIdx.x >> 5;
  const int lane  = threadIdx.x & 31;
  const int npair = blockIdx.x * 8 + wave;   // pair of n-tiles: columns npair*32 .. +31
  const int mtile = blockIdx.y;

  const int half = lane >> 4;      // 0: lanes 0-15, 1: lanes 16-31
  const int r    = lane & 15;      // row (A) / output column (B,C)

  const unsigned short* Ap  = A  + (size_t)(mtile * 16 + r) * lda;
  const unsigned short* Bp0 = Bm + (size_t)(npair * 32 + r) * ldb;
  const unsigned short* Bp1 = Bp0 + (size_t)16 * ldb;

  v8f acc0 = {0.f, 0.f, 0.f, 0.f, 0.f, 0.f, 0.f, 0.f};
  v8f acc1 = {0.f, 0.f, 0.f, 0.f, 0.f, 0.f, 0.f, 0.f};

  for (int k0 = 0; k0 < K; k0 += 32) {
    union { uint4 u[2]; v16bf v; } a, b0, b1;
    // A fragment: elems 0..7 -> K = k0 + half*8 .. ; elems 8..15 -> K = k0+16+half*8 ..
    a.u[0]  = *(const uint4*)(Ap  + k0 + half * 8);
    a.u[1]  = *(const uint4*)(Ap  + k0 + 16 + half * 8);
    // B fragments: lane r holds output column, reading row of W (W[n][k])
    b0.u[0] = *(const uint4*)(Bp0 + k0 + half * 8);
    b0.u[1] = *(const uint4*)(Bp0 + k0 + 16 + half * 8);
    b1.u[0] = *(const uint4*)(Bp1 + k0 + half * 8);
    b1.u[1] = *(const uint4*)(Bp1 + k0 + 16 + half * 8);
    acc0 = __builtin_amdgcn_wmma_f32_16x16x32_bf16(
        false, a.v, false, b0.v, (short)0, acc0, false, false);
    acc1 = __builtin_amdgcn_wmma_f32_16x16x32_bf16(
        false, a.v, false, b1.v, (short)0, acc1, false, false);
  }

  const int col0 = npair * 32 + r;
  const int col1 = col0 + 16;
  const float bv0 = bias ? bias[col0] : 0.f;
  const float bv1 = bias ? bias[col1] : 0.f;
#pragma unroll
  for (int v = 0; v < 8; v++) {
    const int row = mtile * 16 + v + half * 8;       // C/D layout: VGPR v -> M = v + 8*half
    const float val0 = acc0[v] + bv0;
    const float val1 = acc1[v] + bv1;
    C[(size_t)row * ldc + col0] = val0;
    C[(size_t)row * ldc + col1] = val1;
    if (Cbf) {
      Cbf[(size_t)row * ldcbf + col0] = f2bf(val0);
      Cbf[(size_t)row * ldcbf + col1] = f2bf(val1);
    }
  }
}

// ---------------------------------------------------------------- attention step
// scores[b,s] = sum_h tanh(q[b,h] + Wv[b,s,h]) * Vw[h] + bV + mask
// alphas = softmax(scores); ctx[b,h] = sum_s alphas*enc[b,s,h]
// x_bf[b, 0:H] = emb_seq[b,t,:], x_bf[b, H:2H] = bf16(ctx)
__global__ void attn_step_k(const float* __restrict__ q,
                            const float* __restrict__ Wv,
                            const float* __restrict__ Vw,
                            const float* __restrict__ bV,
                            const int* __restrict__ mask,
                            const float* __restrict__ enc,
                            const unsigned short* __restrict__ emb_seq,
                            int t,
                            unsigned short* __restrict__ x_bf,
                            float* __restrict__ alphas_out) {
  const int b = blockIdx.x;
  const int tid = threadIdx.x;
  const int wave = tid >> 5, lane = tid & 31;

  __shared__ float sc[Sq];
  __shared__ float ssum;

  // phase 1: scores (each wave handles 16 s values; lanes split H)
  for (int i = 0; i < 16; i++) {
    const int s = wave * 16 + i;
    const float* wv = Wv + ((size_t)b * Sq + s) * Hq;
    float acc = 0.f;
    for (int h = lane; h < Hq; h += 32)
      acc += tanhf(q[b * Hq + h] + wv[h]) * Vw[h];
#pragma unroll
    for (int off = 16; off > 0; off >>= 1)
      acc += __shfl_xor(acc, off, 32);
    if (lane == 0) {
      float m = (mask[b * Sq + s] == 0) ? -INFINITY : 0.f;
      sc[s] = acc + bV[0] + m;
    }
  }
  __syncthreads();

  // phase 2: softmax over S=128 (wave 0)
  if (tid < 32) {
    float m = -INFINITY;
    for (int s = lane; s < Sq; s += 32) m = fmaxf(m, sc[s]);
#pragma unroll
    for (int off = 16; off > 0; off >>= 1)
      m = fmaxf(m, __shfl_xor(m, off, 32));
    float sum = 0.f;
    for (int s = lane; s < Sq; s += 32) {
      float e = __expf(sc[s] - m);
      sc[s] = e;
      sum += e;
    }
#pragma unroll
    for (int off = 16; off > 0; off >>= 1)
      sum += __shfl_xor(sum, off, 32);
    if (lane == 0) ssum = sum;
  }
  __syncthreads();

  const float inv = 1.f / ssum;
  for (int s = tid; s < Sq; s += 256) {
    float a = sc[s] * inv;
    sc[s] = a;
    alphas_out[((size_t)b * Tq + t) * Sq + s] = a;
  }
  __syncthreads();

  // phase 3: context + assemble x = [emb_t, ctx] in bf16
  for (int h = tid; h < Hq; h += 256) {
    float c = 0.f;
    const float* e = enc + (size_t)b * Sq * Hq + h;
    for (int s = 0; s < Sq; s++) c += sc[s] * e[(size_t)s * Hq];
    x_bf[(size_t)b * 2 * Hq + Hq + h] = f2bf(c);
    x_bf[(size_t)b * 2 * Hq + h] = emb_seq[((size_t)b * Tq + t) * Hq + h];
  }
}

// ---------------------------------------------------------------- GRU gate fusion
// gx,gh: (B,3H) in PyTorch order r,z,n. Updates h in place; writes bf16 copy.
__global__ void gru_gate_k(const float* __restrict__ gx, const float* __restrict__ gh,
                           float* __restrict__ h, unsigned short* __restrict__ h_bf) {
  int i = blockIdx.x * blockDim.x + threadIdx.x;   // over B*H
  if (i >= Bq * Hq) return;
  int hh = i % Hq, b = i / Hq;
  const float xr = gx[b * 3 * Hq + hh];
  const float xz = gx[b * 3 * Hq + Hq + hh];
  const float xn = gx[b * 3 * Hq + 2 * Hq + hh];
  const float hr = gh[b * 3 * Hq + hh];
  const float hz = gh[b * 3 * Hq + Hq + hh];
  const float hn = gh[b * 3 * Hq + 2 * Hq + hh];
  const float r = 1.f / (1.f + __expf(-(xr + hr)));
  const float z = 1.f / (1.f + __expf(-(xz + hz)));
  const float n = tanhf(xn + r * hn);
  const float hv = h[i];
  const float h2 = (1.f - z) * n + z * hv;
  h[i] = h2;
  h_bf[i] = f2bf(h2);
}

// ---------------------------------------------------------------- in-place log_softmax over V
__global__ void log_softmax_k(float* __restrict__ out) {
  const int row = blockIdx.x;                 // b*T + t
  float* p = out + (size_t)row * Vq;
  const int tid = threadIdx.x;
  __shared__ float red[256];

  float m = -INFINITY;
  for (int v = tid; v < Vq; v += 256) m = fmaxf(m, p[v]);
  red[tid] = m; __syncthreads();
  for (int s = 128; s > 0; s >>= 1) {
    if (tid < s) red[tid] = fmaxf(red[tid], red[tid + s]);
    __syncthreads();
  }
  m = red[0]; __syncthreads();

  float sum = 0.f;
  for (int v = tid; v < Vq; v += 256) sum += __expf(p[v] - m);
  red[tid] = sum; __syncthreads();
  for (int s = 128; s > 0; s >>= 1) {
    if (tid < s) red[tid] += red[tid + s];
    __syncthreads();
  }
  const float ls = m + __logf(red[0]);
  __syncthreads();

  for (int v = tid; v < Vq; v += 256) p[v] = p[v] - ls;
}

// ================================================================ host side
static inline size_t alignup(size_t x) { return (x + 255) & ~(size_t)255; }

extern "C" void kernel_launch(void* const* d_in, const int* in_sizes, int n_in,
                              void* d_out, int out_size, void* d_ws, size_t ws_size,
                              hipStream_t stream) {
  (void)in_sizes; (void)n_in; (void)out_size; (void)ws_size;

  const float* enc   = (const float*)d_in[0];   // (B,S,H)
  const float* ench  = (const float*)d_in[1];   // (1,B,H)
  const int*   maskI = (const int*)  d_in[2];   // (B,S)
  const int*   tgt   = (const int*)  d_in[3];   // (B,T)
  const float* emb   = (const float*)d_in[4];   // (V,H)
  const float* W1    = (const float*)d_in[5];
  const float* b1    = (const float*)d_in[6];
  const float* W2    = (const float*)d_in[7];
  const float* b2    = (const float*)d_in[8];
  const float* Vw    = (const float*)d_in[9];
  const float* bV    = (const float*)d_in[10];
  const float* wih   = (const float*)d_in[11];  // (3H,2H)
  const float* whh   = (const float*)d_in[12];  // (3H,H)
  const float* bih   = (const float*)d_in[13];
  const float* bhh   = (const float*)d_in[14];
  const float* Wout  = (const float*)d_in[15];  // (V,H)
  const float* bout  = (const float*)d_in[16];
  const float* Wbr   = (const float*)d_in[17];  // (H,H)
  const float* bbr   = (const float*)d_in[18];

  // ---- workspace carve
  char* p = (char*)d_ws;
  auto carve = [&](size_t bytes) -> void* { void* r = (void*)p; p += alignup(bytes); return r; };

  unsigned short* enc_bf  = (unsigned short*)carve((size_t)Bq * Sq * Hq * 2);
  unsigned short* ench_bf = (unsigned short*)carve((size_t)Bq * Hq * 2);
  unsigned short* Wbr_bf  = (unsigned short*)carve((size_t)Hq * Hq * 2);
  unsigned short* W1_bf   = (unsigned short*)carve((size_t)Hq * Hq * 2);
  unsigned short* W2_bf   = (unsigned short*)carve((size_t)Hq * Hq * 2);
  unsigned short* wih_bf  = (unsigned short*)carve((size_t)3 * Hq * 2 * Hq * 2);
  unsigned short* whh_bf  = (unsigned short*)carve((size_t)3 * Hq * Hq * 2);
  unsigned short* Wout_bf = (unsigned short*)carve((size_t)Vq * Hq * 2);
  unsigned short* embS_bf = (unsigned short*)carve((size_t)Bq * Tq * Hq * 2);
  float*          Wv      = (float*)carve((size_t)Bq * Sq * Hq * 4);
  float*          hbuf    = (float*)carve((size_t)Bq * Hq * 4);
  unsigned short* h_bf    = (unsigned short*)carve((size_t)Bq * Hq * 2);
  float*          qbuf    = (float*)carve((size_t)Bq * Hq * 4);
  unsigned short* x_bf    = (unsigned short*)carve((size_t)Bq * 2 * Hq * 2);
  float*          gx      = (float*)carve((size_t)Bq * 3 * Hq * 4);
  float*          gh      = (float*)carve((size_t)Bq * 3 * Hq * 4);

  float* out        = (float*)d_out;
  float* out_hT     = out + (size_t)Bq * Tq * Vq;
  float* out_alphas = out_hT + (size_t)Bq * Hq;

  auto cast = [&](const float* s, unsigned short* d, int n) {
    cast_f32_bf16_k<<<(n + 255) / 256, 256, 0, stream>>>(s, d, n);
  };
  auto gemm = [&](const unsigned short* A, int lda, const unsigned short* Bm, int ldb,
                  const float* bias, float* C, unsigned long long ldc,
                  unsigned short* Cbf, int ldcbf, int M, int N, int K) {
    dim3 grid(N / 256, M / 16);   // 8 waves * 2 tiles * 16 cols = 256 cols per block
    wmma_gemm_bf16_k<<<grid, 256, 0, stream>>>(A, lda, Bm, ldb, bias, C, ldc, Cbf, ldcbf, M, N, K);
  };

  // ---- one-time preps (re-done every call: deterministic)
  cast(enc,  enc_bf,  Bq * Sq * Hq);
  cast(ench, ench_bf, Bq * Hq);
  cast(Wbr,  Wbr_bf,  Hq * Hq);
  cast(W1,   W1_bf,   Hq * Hq);
  cast(W2,   W2_bf,   Hq * Hq);
  cast(wih,  wih_bf,  3 * Hq * 2 * Hq);
  cast(whh,  whh_bf,  3 * Hq * Hq);
  cast(Wout, Wout_bf, Vq * Hq);
  gather_emb_k<<<(Bq * Tq * Hq + 255) / 256, 256, 0, stream>>>(emb, tgt, embS_bf);

  // Wv = enc @ W2^T + b2      (M = B*S = 4096)
  gemm(enc_bf, Hq, W2_bf, Hq, b2, Wv, Hq, nullptr, 0, Bq * Sq, Hq, Hq);
  // h0 = enc_hidden @ Wbr^T + bbr  (also writes bf16 copy for next GEMMs)
  gemm(ench_bf, Hq, Wbr_bf, Hq, bbr, hbuf, Hq, h_bf, Hq, Bq, Hq, Hq);

  // ---- decode loop
  for (int t = 0; t < Tq; t++) {
    // q = h @ W1^T + b1
    gemm(h_bf, Hq, W1_bf, Hq, b1, qbuf, Hq, nullptr, 0, Bq, Hq, Hq);
    // attention: scores -> softmax -> ctx, assemble x_bf
    attn_step_k<<<Bq, 256, 0, stream>>>(qbuf, Wv, Vw, bV, maskI, enc, embS_bf, t, x_bf, out_alphas);
    // gx = x @ w_ih^T + b_ih   (K = 2H)
    gemm(x_bf, 2 * Hq, wih_bf, 2 * Hq, bih, gx, 3 * Hq, nullptr, 0, Bq, 3 * Hq, 2 * Hq);
    // gh = h @ w_hh^T + b_hh   (K = H)
    gemm(h_bf, Hq, whh_bf, Hq, bhh, gh, 3 * Hq, nullptr, 0, Bq, 3 * Hq, Hq);
    // GRU gates -> new h (and bf16 copy)
    gru_gate_k<<<(Bq * Hq + 255) / 256, 256, 0, stream>>>(gx, gh, hbuf, h_bf);
    // logits = h2 @ Wout^T + bout, written strided into d_out at [:, t, :]
    gemm(h_bf, Hq, Wout_bf, Hq, bout, out + (size_t)t * Vq,
         (unsigned long long)Tq * Vq, nullptr, 0, Bq, Vq, Hq);
  }

  // ---- finalize: in-place log_softmax over vocab, copy hT
  log_softmax_k<<<Bq * Tq, 256, 0, stream>>>(out);
  hipMemcpyAsync(out_hT, hbuf, (size_t)Bq * Hq * sizeof(float),
                 hipMemcpyDeviceToDevice, stream);
}
